// MultiScaleBiMamba_75883482186474
// MI455X (gfx1250) — matrix operand order repaired and checked by reading, not compiled, not run on hardware
//
#include <hip/hip_runtime.h>
#include <hip/hip_bf16.h>
#include <math.h>

// ---------------- model constants ----------------
#define DMC 768
#define DSC 256
#define DIC 512
#define NST 16
#define KCV 4
#define RRK 16
#define BB  8
#define LL  1024
#define MROWS (BB * LL)   // 8192
#define BN_EPS 1e-5f

// ---------------- WMMA types ----------------
typedef __attribute__((ext_vector_type(16))) __bf16 v16bf;
typedef __attribute__((ext_vector_type(8)))  float  v8f;

union AFrag { v16bf v; unsigned u[8]; };

__device__ __forceinline__ unsigned short f32_to_bf16_rne(float f) {
    unsigned u = __float_as_uint(f);
    u += 0x7FFFu + ((u >> 16) & 1u);
    return (unsigned short)(u >> 16);
}
__device__ __forceinline__ unsigned pack_bf16x2(float lo, float hi) {
    return ((unsigned)f32_to_bf16_rne(hi) << 16) | (unsigned)f32_to_bf16_rne(lo);
}

__device__ __forceinline__ float siluf(float v)  { return v / (1.f + __expf(-v)); }
__device__ __forceinline__ float sigmoidf_(float v) { return 1.f / (1.f + __expf(-v)); }
__device__ __forceinline__ float softplusf_(float v) {
    return (v > 20.f) ? v : log1pf(__expf(v));
}
__device__ __forceinline__ float gelu_exact(float v) {
    return 0.5f * v * (1.f + erff(v * 0.70710678118654752f));
}

// ============================================================================
// bf16 WMMA GEMM:  C[M,N] = A[M,K] * W[N,K]^T (+ bias)
//   block = 256 threads = 8 waves (4 along M x 2 along N)
//   block tile 128(M) x 64(N) x 32(K); each wave owns a 32x32 sub-tile
//   -> 4 x v_wmma_f32_16x16x32_bf16 per wave per K-step.
// Requirements: M % 128 == 0, Kd % 32 == 0 (all launches satisfy this);
//               Nout guarded (handles Nout=48).
// ============================================================================
#define TM 128
#define TN 64
#define TK 32
#define STR 40   // shorts per LDS row: 80B (8B aligned, conflict-free reads)

template <bool HAS_BIAS>
__global__ __launch_bounds__(256) void gemm_bf16_wmma(
    const float* __restrict__ A, int lda,
    const float* __restrict__ W, int ldw,      // W row-major [Nout][Kd], ldw == Kd
    const float* __restrict__ bias,
    float* __restrict__ C, int ldc,
    int M, int Nout, int Kd)
{
    __shared__ __align__(16) unsigned short As[TM * STR];
    __shared__ __align__(16) unsigned short Bs[TN * STR];

    const int tid  = threadIdx.x;
    const int lane = tid & 31;
    const int wave = tid >> 5;
    const int wm   = wave & 3;       // 0..3 -> 32-row slab
    const int wn   = wave >> 2;      // 0..1 -> 32-col slab
    const int blockM = blockIdx.y * TM;
    const int blockN = blockIdx.x * TN;
    const int row16 = lane & 15;
    const int half  = lane >> 4;     // K-half select per ISA fragment layout

    // per-thread staging coordinates (float4 granularity, 8 float4 per 32-K row)
    const int aRow0 = tid >> 3;          // + i*32 rows, i=0..3  -> 128 rows
    const int k4    = (tid & 7) * 4;     // K offset of this thread's float4
    const int bRow0 = tid >> 3;          // + i*32 rows, i=0..1  -> 64 rows

    v8f acc00 = {}, acc01 = {}, acc10 = {}, acc11 = {};

    for (int k0 = 0; k0 < Kd; k0 += TK) {
        // ---- global -> regs (overlaps with previous tile's wmmas) ----
        float4 aReg[4];
#pragma unroll
        for (int i = 0; i < 4; ++i) {
            aReg[i] = *(const float4*)&A[(size_t)(blockM + aRow0 + i * 32) * lda + k0 + k4];
        }
        float4 bReg[2];
#pragma unroll
        for (int i = 0; i < 2; ++i) {
            int gn  = blockN + bRow0 + i * 32;
            int gnc = gn < Nout ? gn : Nout - 1;            // clamp: branch-free
            float4 v = *(const float4*)&W[(size_t)gnc * ldw + k0 + k4];
            if (gn >= Nout) { v.x = 0.f; v.y = 0.f; v.z = 0.f; v.w = 0.f; }
            bReg[i] = v;
        }
        if (k0 + TK < Kd)  // prefetch next A tile (global_prefetch_b8 path)
            __builtin_prefetch(&A[(size_t)(blockM + (tid >> 1)) * lda + k0 + TK], 0, 0);

        __syncthreads();   // previous tile's fragment reads complete

        // ---- regs -> LDS as packed bf16 (ds_store_b64) ----
#pragma unroll
        for (int i = 0; i < 4; ++i) {
            uint2 p; p.x = pack_bf16x2(aReg[i].x, aReg[i].y);
                     p.y = pack_bf16x2(aReg[i].z, aReg[i].w);
            *(uint2*)(void*)&As[(aRow0 + i * 32) * STR + k4] = p;
        }
#pragma unroll
        for (int i = 0; i < 2; ++i) {
            uint2 p; p.x = pack_bf16x2(bReg[i].x, bReg[i].y);
                     p.y = pack_bf16x2(bReg[i].z, bReg[i].w);
            *(uint2*)(void*)&Bs[(bRow0 + i * 32) * STR + k4] = p;
        }
        __syncthreads();

        // ---- fragments per ISA 7.12.2 layouts ----
        AFrag a[2], b[2];
#pragma unroll
        for (int mi = 0; mi < 2; ++mi) {
            const unsigned short* aR = &As[(wm * 32 + mi * 16 + row16) * STR];
#pragma unroll
            for (int v = 0; v < 8; ++v) {
                // A 16x32 bf16: VGPR v<4: K=half*8+2v ; v>=4: K=16+half*8+2(v-4)
                int kl = (v < 4) ? (half * 8 + 2 * v) : (16 + half * 8 + 2 * (v - 4));
                a[mi].u[v] = *(const unsigned*)(const void*)(aR + kl);
            }
        }
#pragma unroll
        for (int ni = 0; ni < 2; ++ni) {
            const unsigned short* bR = &Bs[(wn * 32 + ni * 16 + row16) * STR];
#pragma unroll
            for (int v = 0; v < 8; ++v) {
                // B 32x16 bf16: VGPR v<4: K=half*8+2v ; v>=4: K=16+half*8+2(v-4)
                int kl = (v < 4) ? (half * 8 + 2 * v) : (16 + half * 8 + 2 * (v - 4));
                b[ni].u[v] = *(const unsigned*)(const void*)(bR + kl);
            }
        }

        acc00 = __builtin_amdgcn_wmma_f32_16x16x32_bf16(false, a[0].v, false, b[0].v, (short)0, acc00, false, false);
        acc01 = __builtin_amdgcn_wmma_f32_16x16x32_bf16(false, a[0].v, false, b[1].v, (short)0, acc01, false, false);
        acc10 = __builtin_amdgcn_wmma_f32_16x16x32_bf16(false, a[1].v, false, b[0].v, (short)0, acc10, false, false);
        acc11 = __builtin_amdgcn_wmma_f32_16x16x32_bf16(false, a[1].v, false, b[1].v, (short)0, acc11, false, false);
        // next iteration's pre-store barrier protects LDS against overwrite
    }

    // ---- epilogue: C/D layout (VGPR r -> M = half*8 + r ; N = lane&15) ----
    const int col0 = blockN + wn * 32 + row16;
    const int col1 = col0 + 16;
    const float bv0 = (HAS_BIAS && col0 < Nout) ? bias[col0] : 0.f;
    const float bv1 = (HAS_BIAS && col1 < Nout) ? bias[col1] : 0.f;
#pragma unroll
    for (int r = 0; r < 8; ++r) {
        int row0 = blockM + wm * 32 + half * 8 + r;
        int row1 = row0 + 16;
        if (col0 < Nout) {
            C[(size_t)row0 * ldc + col0] = acc00[r] + bv0;
            C[(size_t)row1 * ldc + col0] = acc10[r] + bv0;
        }
        if (col1 < Nout) {
            C[(size_t)row0 * ldc + col1] = acc01[r] + bv1;
            C[(size_t)row1 * ldc + col1] = acc11[r] + bv1;
        }
    }
}

// ---------------- lengths = mask.sum(1) ----------------
__global__ void lengths_kernel(const int* __restrict__ mask, int* __restrict__ lengths) {
    int b = threadIdx.x;
    if (b >= BB) return;
    int s = 0;
    for (int l = 0; l < LL; ++l) s += mask[b * LL + l];
    lengths[b] = s;
}

// ---------------- dilated depthwise conv (k=3, SAME) + BN + GELU ----------------
__global__ void conv_bn_gelu_kernel(
    const float* __restrict__ hidden,
    const float* __restrict__ pw, const float* __restrict__ pb,
    const float* __restrict__ g, const float* __restrict__ bb_,
    const float* __restrict__ rm, const float* __restrict__ rv,
    float* __restrict__ c, int dil, int chOff)
{
    size_t tid = (size_t)blockIdx.x * blockDim.x + threadIdx.x;
    if (tid >= (size_t)BB * LL * DSC) return;
    int ch = tid % DSC;
    int l  = (tid / DSC) % LL;
    int b  = tid / ((size_t)DSC * LL);

    float acc = pb[ch];
    const float* w = &pw[ch * 3];
#pragma unroll
    for (int j = 0; j < 3; ++j) {
        int ll = l + (j - 1) * dil;
        if (ll >= 0 && ll < LL)
            acc += w[j] * hidden[((size_t)b * LL + ll) * DMC + chOff + ch];
    }
    acc = (acc - rm[ch]) * rsqrtf(rv[ch] + BN_EPS) * g[ch] + bb_[ch];
    c[((size_t)b * LL + l) * DSC + ch] = gelu_exact(acc);
}

// ---------------- masked flip (strided) ----------------
__global__ void flip_kernel(const float* __restrict__ src, int lds_,
                            float* __restrict__ dst, int ldd,
                            const int* __restrict__ lengths, int Cc)
{
    size_t tid = (size_t)blockIdx.x * blockDim.x + threadIdx.x;
    if (tid >= (size_t)BB * LL * Cc) return;
    int cc = tid % Cc;
    int l  = (tid / Cc) % LL;
    int b  = tid / ((size_t)Cc * LL);
    int len = lengths[b];
    int idx = (l < len) ? (len - 1 - l) : l;
    dst[((size_t)b * LL + l) * ldd + cc] = src[((size_t)b * LL + idx) * lds_ + cc];
}

// ---------------- causal depthwise conv (K=4) + SiLU on x half of xz ----------------
__global__ void causal_conv_silu_kernel(
    const float* __restrict__ xz, const float* __restrict__ cw,
    const float* __restrict__ cb, float* __restrict__ xc)
{
    size_t tid = (size_t)blockIdx.x * blockDim.x + threadIdx.x;
    if (tid >= (size_t)BB * LL * DIC) return;
    int d = tid % DIC;
    int l = (tid / DIC) % LL;
    int b = tid / ((size_t)DIC * LL);

    float acc = cb[d];
    const float* w = &cw[d * KCV];
#pragma unroll
    for (int j = 0; j < KCV; ++j) {
        int ll = l - (KCV - 1) + j;
        if (ll >= 0)
            acc += w[j] * xz[((size_t)b * LL + ll) * (2 * DIC) + d];
    }
    xc[((size_t)b * LL + l) * DIC + d] = siluf(acc);
}

// ---------------- dt = softplus(xdbl[:, :R] @ dw^T + db) ----------------
__global__ void dt_softplus_kernel(
    const float* __restrict__ xdbl, const float* __restrict__ dw,
    const float* __restrict__ db, float* __restrict__ dt)
{
    size_t tid = (size_t)blockIdx.x * blockDim.x + threadIdx.x;
    if (tid >= (size_t)MROWS * DIC) return;
    int d = tid % DIC;
    size_t m = tid / DIC;
    const float* xr = &xdbl[m * (RRK + 2 * NST)];
    const float* wr = &dw[d * RRK];
    float acc = db[d];
#pragma unroll
    for (int r = 0; r < RRK; ++r) acc += xr[r] * wr[r];
    dt[m * DIC + d] = softplusf_(acc);
}

// ---------------- selective scan + skip + SiLU(z) gate ----------------
__global__ void scan_kernel(
    const float* __restrict__ xc, const float* __restrict__ dt,
    const float* __restrict__ xdbl, const float* __restrict__ alog,
    const float* __restrict__ Dp, const float* __restrict__ xz,
    float* __restrict__ u2)
{
    int tid = blockIdx.x * blockDim.x + threadIdx.x;
    if (tid >= BB * DIC) return;
    int d = tid % DIC;
    int b = tid / DIC;

    float Arow[NST], h[NST];
#pragma unroll
    for (int n = 0; n < NST; ++n) { Arow[n] = -__expf(alog[d * NST + n]); h[n] = 0.f; }
    float Dd = Dp[d];

    for (int t = 0; t < LL; ++t) {
        size_t base = (size_t)b * LL + t;
        float x   = xc[base * DIC + d];
        float dtv = dt[base * DIC + d];
        float z   = xz[base * (2 * DIC) + DIC + d];
        const float* xd = &xdbl[base * (RRK + 2 * NST)];
        float dx = dtv * x;
        float y = 0.f;
#pragma unroll
        for (int n = 0; n < NST; ++n) {
            float Bn = xd[RRK + n];
            float Cn = xd[RRK + NST + n];
            h[n] = h[n] * __expf(dtv * Arow[n]) + dx * Bn;
            y += h[n] * Cn;
        }
        y += x * Dd;
        u2[base * DIC + d] = y * siluf(z);
    }
}

// ---------------- mean / max over L ----------------
__global__ void reduce_kernel(const float* __restrict__ outs,
                              float* __restrict__ avg, float* __restrict__ mx)
{
    int tid = blockIdx.x * blockDim.x + threadIdx.x;
    if (tid >= BB * DMC) return;
    int c = tid % DMC, b = tid / DMC;
    float s = 0.f, m = -3.402823466e+38f;
    for (int l = 0; l < LL; ++l) {
        float v = outs[((size_t)b * LL + l) * DMC + c];
        s += v; m = fmaxf(m, v);
    }
    avg[b * DMC + c] = s / (float)LL;
    mx[b * DMC + c]  = m;
}

// ---------------- attention MLP layer 1 ----------------
__global__ void mlp1_kernel(const float* __restrict__ avg, const float* __restrict__ mx,
                            const float* __restrict__ w1,
                            float* __restrict__ ha, float* __restrict__ hm)
{
    int tid = blockIdx.x * blockDim.x + threadIdx.x;
    if (tid >= 2 * BB * (DMC / 2)) return;
    int sel = tid / (BB * (DMC / 2));
    int r   = tid % (BB * (DMC / 2));
    int b = r / (DMC / 2), j = r % (DMC / 2);
    const float* vin = sel ? &mx[b * DMC] : &avg[b * DMC];
    const float* wr  = &w1[j * DMC];
    float acc = 0.f;
    for (int c = 0; c < DMC; ++c) acc += vin[c] * wr[c];
    float* o = sel ? hm : ha;
    o[b * (DMC / 2) + j] = fmaxf(acc, 0.f);
}

// ---------------- att = sigmoid( (ha+hm) @ W2^T ) ----------------
__global__ void att_kernel(const float* __restrict__ ha, const float* __restrict__ hm,
                           const float* __restrict__ w2, float* __restrict__ att)
{
    int tid = blockIdx.x * blockDim.x + threadIdx.x;
    if (tid >= BB * DMC) return;
    int c = tid % DMC, b = tid / DMC;
    const float* wr = &w2[c * (DMC / 2)];
    float acc = 0.f;
    for (int j = 0; j < DMC / 2; ++j)
        acc += (ha[b * (DMC / 2) + j] + hm[b * (DMC / 2) + j]) * wr[j];
    att[b * DMC + c] = sigmoidf_(acc);
}

// ---------------- outs *= att (broadcast over L) ----------------
__global__ void scale_kernel(float* __restrict__ outs, const float* __restrict__ att)
{
    size_t tid = (size_t)blockIdx.x * blockDim.x + threadIdx.x;
    if (tid >= (size_t)BB * LL * DMC) return;
    int c = tid % DMC;
    int b = tid / ((size_t)DMC * LL);
    outs[tid] *= att[b * DMC + c];
}

// ---------------- orchestration ----------------
extern "C" void kernel_launch(void* const* d_in, const int* in_sizes, int n_in,
                              void* d_out, int out_size, void* d_ws, size_t ws_size,
                              hipStream_t stream) {
    const float* hidden  = (const float*)d_in[0];
    const int*   mask    = (const int*)d_in[1];
    const float* pool_w  = (const float*)d_in[2];
    const float* pool_b  = (const float*)d_in[3];
    const float* bn_g    = (const float*)d_in[4];
    const float* bn_b    = (const float*)d_in[5];
    const float* bn_rm   = (const float*)d_in[6];
    const float* bn_rv   = (const float*)d_in[7];
    const float* in_proj = (const float*)d_in[8];
    const float* conv_w  = (const float*)d_in[9];
    const float* conv_b  = (const float*)d_in[10];
    const float* x_proj  = (const float*)d_in[11];
    const float* dt_w    = (const float*)d_in[12];
    const float* dt_b    = (const float*)d_in[13];
    const float* A_log   = (const float*)d_in[14];
    const float* Dp      = (const float*)d_in[15];
    const float* mo_w    = (const float*)d_in[16];
    const float* op_w    = (const float*)d_in[17];
    const float* op_b    = (const float*)d_in[18];
    const float* ca_w1   = (const float*)d_in[19];
    const float* ca_w2   = (const float*)d_in[20];
    const float* fus_w   = (const float*)d_in[21];
    const float* fus_b   = (const float*)d_in[22];
    float* out = (float*)d_out;

    float* ws = (float*)d_ws;
    size_t o = 0;
    float* cbuf  = ws + o; o += (size_t)MROWS * DSC;
    float* cflip = ws + o; o += (size_t)MROWS * DSC;
    float* xz    = ws + o; o += (size_t)MROWS * 2 * DIC;
    float* xc    = ws + o; o += (size_t)MROWS * DIC;
    float* xdbl  = ws + o; o += (size_t)MROWS * (RRK + 2 * NST);
    float* dtb   = ws + o; o += (size_t)MROWS * DIC;
    float* u2    = ws + o; o += (size_t)MROWS * DIC;
    float* cat   = ws + o; o += (size_t)MROWS * 2 * DSC;
    float* bwdt  = ws + o; o += (size_t)MROWS * DSC;
    float* outs  = ws + o; o += (size_t)MROWS * DMC;
    float* avg   = ws + o; o += BB * DMC;
    float* mx    = ws + o; o += BB * DMC;
    float* ha    = ws + o; o += BB * (DMC / 2);
    float* hm    = ws + o; o += BB * (DMC / 2);
    float* att   = ws + o; o += BB * DMC;
    int* lengths = (int*)(ws + o);

    const int scales[3] = {1, 2, 4};

    lengths_kernel<<<1, 32, 0, stream>>>(mask, lengths);

    const size_t nBLDS = (size_t)BB * LL * DSC;
    const size_t nBLDI = (size_t)BB * LL * DIC;
    const size_t nBLDM = (size_t)BB * LL * DMC;
    const int gridM = MROWS / TM;    // 64

    for (int i = 0; i < 3; ++i) {
        conv_bn_gelu_kernel<<<(nBLDS + 255) / 256, 256, 0, stream>>>(
            hidden, pool_w + (size_t)i * DSC * 3, pool_b + (size_t)i * DSC,
            bn_g + (size_t)i * DSC, bn_b + (size_t)i * DSC,
            bn_rm + (size_t)i * DSC, bn_rv + (size_t)i * DSC,
            cbuf, scales[i], i * DSC);
        flip_kernel<<<(nBLDS + 255) / 256, 256, 0, stream>>>(
            cbuf, DSC, cflip, DSC, lengths, DSC);

        for (int dir = 0; dir < 2; ++dir) {
            const size_t sd = (size_t)(i * 2 + dir);
            const float* cin = dir ? cflip : cbuf;

            // in_proj: [8192,256] @ [1024,256]^T -> xz
            gemm_bf16_wmma<false><<<dim3((2 * DIC + TN - 1) / TN, gridM), 256, 0, stream>>>(
                cin, DSC, in_proj + sd * (2 * DIC) * DSC, DSC, nullptr,
                xz, 2 * DIC, MROWS, 2 * DIC, DSC);

            causal_conv_silu_kernel<<<(nBLDI + 255) / 256, 256, 0, stream>>>(
                xz, conv_w + sd * DIC * KCV, conv_b + sd * DIC, xc);

            // x_proj: [8192,512] @ [48,512]^T -> xdbl
            gemm_bf16_wmma<false><<<dim3(1, gridM), 256, 0, stream>>>(
                xc, DIC, x_proj + sd * (RRK + 2 * NST) * DIC, DIC, nullptr,
                xdbl, RRK + 2 * NST, MROWS, RRK + 2 * NST, DIC);

            dt_softplus_kernel<<<((size_t)MROWS * DIC + 255) / 256, 256, 0, stream>>>(
                xdbl, dt_w + sd * DIC * RRK, dt_b + sd * DIC, dtb);

            scan_kernel<<<(BB * DIC + 255) / 256, 256, 0, stream>>>(
                xc, dtb, xdbl, A_log + sd * DIC * NST, Dp + sd * DIC, xz, u2);

            if (dir == 0) {
                gemm_bf16_wmma<false><<<dim3(DSC / TN, gridM), 256, 0, stream>>>(
                    u2, DIC, mo_w + sd * DSC * DIC, DIC, nullptr,
                    cat, 2 * DSC, MROWS, DSC, DIC);
            } else {
                gemm_bf16_wmma<false><<<dim3(DSC / TN, gridM), 256, 0, stream>>>(
                    u2, DIC, mo_w + sd * DSC * DIC, DIC, nullptr,
                    bwdt, DSC, MROWS, DSC, DIC);
                flip_kernel<<<(nBLDS + 255) / 256, 256, 0, stream>>>(
                    bwdt, DSC, cat + DSC, 2 * DSC, lengths, DSC);
            }
        }

        gemm_bf16_wmma<true><<<dim3(DSC / TN, gridM), 256, 0, stream>>>(
            cat, 2 * DSC, op_w + (size_t)i * DSC * 2 * DSC, 2 * DSC,
            op_b + (size_t)i * DSC, outs + i * DSC, DMC, MROWS, DSC, 2 * DSC);
    }

    reduce_kernel<<<(BB * DMC + 255) / 256, 256, 0, stream>>>(outs, avg, mx);
    mlp1_kernel<<<(2 * BB * (DMC / 2) + 255) / 256, 256, 0, stream>>>(avg, mx, ca_w1, ha, hm);
    att_kernel<<<(BB * DMC + 255) / 256, 256, 0, stream>>>(ha, hm, ca_w2, att);
    scale_kernel<<<(nBLDM + 255) / 256, 256, 0, stream>>>(outs, att);

    gemm_bf16_wmma<true><<<dim3(DMC / TN, gridM), 256, 0, stream>>>(
        outs, DMC, fus_w, DMC, fus_b, out, DMC, MROWS, DMC, DMC);

    (void)in_sizes; (void)n_in; (void)out_size; (void)ws_size;
}
